// GNN_Bet_49873160241783
// MI455X (gfx1250) — matrix operand order repaired and checked by reading, not compiled, not run on hardware
//
#include <hip/hip_runtime.h>
#include <hip/hip_bf16.h>

typedef _Float16 half4v __attribute__((ext_vector_type(4)));
typedef _Float16 half8v __attribute__((ext_vector_type(8)));
typedef _Float16 v16h   __attribute__((ext_vector_type(16)));
typedef float    v8f    __attribute__((ext_vector_type(8)));

#define NHID 128
#define NH2  256

// ---------------- weight prep: f32 [R,C] -> f16 transposed [C,R] ----------------
__global__ void k_cvt_transpose_f16(const float* __restrict__ in, _Float16* __restrict__ out,
                                    int R, int C) {
  int t = blockIdx.x * blockDim.x + threadIdx.x;
  if (t < R * C) {
    int r = t / C, c = t % C;
    out[(size_t)c * R + r] = (_Float16)in[(size_t)r * C + c];
  }
}

// ---------------- elementwise f32 -> f16 ----------------
__global__ void k_cvt_f16(const float* __restrict__ in, _Float16* __restrict__ out, size_t n) {
  size_t t = blockIdx.x * (size_t)blockDim.x + threadIdx.x;
  if (t < n) out[t] = (_Float16)in[t];
}

// ================= CSR construction (per adjacency, reused for 6 SpMMs) =================
__global__ void k_hist(const int* __restrict__ dst, int* __restrict__ counts, int E) {
  int t = blockIdx.x * blockDim.x + threadIdx.x;
  if (t < E)
    __hip_atomic_fetch_add(&counts[dst[t]], 1, __ATOMIC_RELAXED, __HIP_MEMORY_SCOPE_AGENT);
}

// single-block hierarchical exclusive scan over N counts -> offsets[N+1]
__global__ void k_scan(const int* __restrict__ counts, int* __restrict__ offsets, int N) {
  __shared__ int warpsum[32];
  __shared__ int carry;
  if (threadIdx.x == 0) carry = 0;
  int lane = threadIdx.x & 31, wid = threadIdx.x >> 5;
  for (int base = 0; base < N; base += 1024) {
    int i = base + threadIdx.x;
    int v = (i < N) ? counts[i] : 0;
    int sv = v;                                   // inclusive warp scan
    for (int d = 1; d < 32; d <<= 1) { int n = __shfl_up(sv, d, 32); if (lane >= d) sv += n; }
    __syncthreads();                              // warpsum free to overwrite, carry visible
    if (lane == 31) warpsum[wid] = sv;
    __syncthreads();
    if (wid == 0) {
      int ws = warpsum[lane];
      for (int d = 1; d < 32; d <<= 1) { int n = __shfl_up(ws, d, 32); if (lane >= d) ws += n; }
      warpsum[lane] = ws;                         // inclusive scan of warp totals
    }
    __syncthreads();
    int excl = sv - v + (wid > 0 ? warpsum[wid - 1] : 0) + carry;
    if (i < N) offsets[i] = excl;
    __syncthreads();
    if (threadIdx.x == 0) carry += warpsum[31];   // chunk total
  }
  __syncthreads();
  if (threadIdx.x == 0) offsets[N] = carry;
}

__global__ void k_copy_int(const int* __restrict__ in, int* __restrict__ out, int N) {
  int t = blockIdx.x * blockDim.x + threadIdx.x;
  if (t < N) out[t] = in[t];
}

__global__ void k_scatter(const int* __restrict__ src, const int* __restrict__ dst,
                          const float* __restrict__ w, int* __restrict__ cursor,
                          int* __restrict__ csr_src, float* __restrict__ csr_w, int E) {
  int t = blockIdx.x * blockDim.x + threadIdx.x;
  if (t < E) {
    int pos = __hip_atomic_fetch_add(&cursor[dst[t]], 1, __ATOMIC_RELAXED,
                                     __HIP_MEMORY_SCOPE_AGENT);
    csr_src[pos] = src[t];
    csr_w[pos]   = w[t];
  }
}

// ---------------- SpMM over CSR: one wave per destination node, no atomics ----------------
__global__ void k_spmm_csr(const int* __restrict__ off, const int* __restrict__ csr_src,
                           const float* __restrict__ csr_w, const float* __restrict__ x,
                           float* __restrict__ y, int N) {
  unsigned tid = blockIdx.x * (unsigned)blockDim.x + threadIdx.x;
  int row = (int)(tid >> 5);
  int lane = threadIdx.x & 31;
  if (row >= N) return;
  int beg = off[row], end = off[row + 1];
  float4 acc = {0.f, 0.f, 0.f, 0.f};
  for (int j0 = beg; j0 < end; j0 += 32) {
    int jj = j0 + lane;
    int   sv = 0;
    float wv = 0.f;
    if (jj < end) { sv = csr_src[jj]; wv = csr_w[jj]; }
    int cnt = end - j0; if (cnt > 32) cnt = 32;   // wave-uniform
    for (int t = 0; t < cnt; ++t) {
      int   s  = __shfl(sv, t, 32);
      float we = __shfl(wv, t, 32);
      float4 v = ((const float4*)(x + (size_t)s * NHID))[lane];
      acc.x += we * v.x; acc.y += we * v.y; acc.z += we * v.z; acc.w += we * v.w;
    }
  }
  ((float4*)(y + (size_t)row * NHID))[lane] = acc;
}

// ---------------- bias + relu + (optional) row l2norm; emit f16 + accumulate x7 ----------------
__global__ void k_post(const float* __restrict__ y, const float* __restrict__ bias,
                       _Float16* __restrict__ ah, float* __restrict__ x7sum,
                       int N, int do_norm) {
  unsigned tid = blockIdx.x * (unsigned)blockDim.x + threadIdx.x;
  int row = (int)(tid >> 5);
  int lane = threadIdx.x & 31;
  if (row >= N) return;
  float4 v = ((const float4*)(y + (size_t)row * NHID))[lane];
  float4 b = ((const float4*)bias)[lane];
  v.x = fmaxf(v.x + b.x, 0.f);
  v.y = fmaxf(v.y + b.y, 0.f);
  v.z = fmaxf(v.z + b.z, 0.f);
  v.w = fmaxf(v.w + b.w, 0.f);
  if (do_norm) {
    float ss = v.x * v.x + v.y * v.y + v.z * v.z + v.w * v.w;
    for (int m = 16; m >= 1; m >>= 1) ss += __shfl_xor(ss, m, 32);
    float scale = 1.f / fmaxf(sqrtf(ss), 1e-12f);
    v.x *= scale; v.y *= scale; v.z *= scale; v.w *= scale;
  }
  float4* xs = (float4*)(x7sum + (size_t)row * NHID) + lane;
  float4 acc = *xs;
  acc.x += v.x; acc.y += v.y; acc.z += v.z; acc.w += v.w;
  *xs = acc;
  half4v h = { (_Float16)v.x, (_Float16)v.y, (_Float16)v.z, (_Float16)v.w };
  *((half4v*)(ah + (size_t)row * NHID) + lane) = h;
}

// ---------------- WMMA GEMM: C[M,Nc] = A[M,K] @ Bt[Nc,K]^T, fused bias/relu ----------------
// One wave -> 16x64 C region (1 A fragment reused across 4 N-tiles, 4 accumulators).
__global__ void k_gemm_f16(const _Float16* __restrict__ A, const _Float16* __restrict__ Bt,
                           int M, int K, int Nc,
                           const float* __restrict__ bias, int relu,
                           float* __restrict__ Cf, _Float16* __restrict__ Ch) {
  unsigned tid = blockIdx.x * (unsigned)blockDim.x + threadIdx.x;
  int wave = (int)(tid >> 5);
  int lane = threadIdx.x & 31;
  int tilesG = Nc >> 6;                 // groups of 4 N-tiles (64 cols)
  int tm = wave / tilesG;
  int tg = wave - tm * tilesG;
  int Mt = (M + 15) >> 4;
  if (tm >= Mt) return;

  int g = lane >> 4;                    // half-wave group
  int l = lane & 15;
  int arow = tm * 16 + l; if (arow > M - 1) arow = M - 1;
  // A 16x32 f16 fragment: group 0 holds K 0..7 & 16..23, group 1 holds K 8..15 & 24..31
  const _Float16* Ap = A + (size_t)arow * K + g * 8;
  // B 32x16 f16 fragment: group 0 holds K 0..7 & 16..23, group 1 holds K 8..15 & 24..31 (column of B)
  const _Float16* Bp0 = Bt + (size_t)(tg * 64 + l) * K + g * 8;

  v8f acc[4];
#pragma unroll
  for (int j = 0; j < 4; ++j) acc[j] = (v8f){0.f, 0.f, 0.f, 0.f, 0.f, 0.f, 0.f, 0.f};

#pragma unroll 4
  for (int k = 0; k < K; k += 32) {
    half8v alo = *(const half8v*)(Ap + k);
    half8v ahi = *(const half8v*)(Ap + k + 16);
    v16h a = __builtin_shufflevector(alo, ahi, 0,1,2,3,4,5,6,7,8,9,10,11,12,13,14,15);
#pragma unroll
    for (int j = 0; j < 4; ++j) {
      const _Float16* Bp = Bp0 + (size_t)(j * 16) * K;
      half8v blo = *(const half8v*)(Bp + k);
      half8v bhi = *(const half8v*)(Bp + k + 16);
      v16h bm = __builtin_shufflevector(blo, bhi, 0,1,2,3,4,5,6,7,8,9,10,11,12,13,14,15);
      acc[j] = __builtin_amdgcn_wmma_f32_16x16x32_f16(false, a, false, bm, (short)0, acc[j],
                                                      false, false);
    }
  }

  int rbase = tm * 16 + g * 8;          // C layout: vgpr r -> row rbase+r, col = lane&15
#pragma unroll
  for (int j = 0; j < 4; ++j) {
    int col = tg * 64 + j * 16 + l;
    float bcol = bias ? bias[col] : 0.f;
#pragma unroll
    for (int r = 0; r < 8; r++) {
      int row = rbase + r;
      if (row < M) {
        float v = acc[j][r] + bcol;
        if (relu) v = fmaxf(v, 0.f);
        if (Cf) Cf[(size_t)row * Nc + col] = v;
        else    Ch[(size_t)row * Nc + col] = (_Float16)v;
      }
    }
  }
}

// ---------------- s[row] += h2[row,:256] . L3 + bl3 ----------------
__global__ void k_dot_l3(const _Float16* __restrict__ h2, const float* __restrict__ L3,
                         const float* __restrict__ bl3, float* __restrict__ s, int N) {
  unsigned tid = blockIdx.x * (unsigned)blockDim.x + threadIdx.x;
  int row = (int)(tid >> 5);
  int lane = threadIdx.x & 31;
  if (row >= N) return;
  half8v h = *((const half8v*)(h2 + (size_t)row * NH2) + lane);
  const float4* lp = (const float4*)L3 + lane * 2;
  float4 l0 = lp[0], l1 = lp[1];
  float d = (float)h[0] * l0.x + (float)h[1] * l0.y + (float)h[2] * l0.z + (float)h[3] * l0.w
          + (float)h[4] * l1.x + (float)h[5] * l1.y + (float)h[6] * l1.z + (float)h[7] * l1.w;
  for (int m = 16; m >= 1; m >>= 1) d += __shfl_xor(d, m, 32);
  if (lane == 0) s[row] += d + bl3[0];
}

// ---------------- score = s/7 ; out = score (branch 0) or out *= score (branch 1) ----------------
__global__ void k_finalize(const float* __restrict__ s, float* __restrict__ out, int N, int mul) {
  int t = blockIdx.x * blockDim.x + threadIdx.x;
  if (t < N) {
    float v = s[t] * (1.f / 7.f);
    out[t] = mul ? out[t] * v : v;
  }
}

extern "C" void kernel_launch(void* const* d_in, const int* in_sizes, int n_in,
                              void* d_out, int out_size, void* d_ws, size_t ws_size,
                              hipStream_t stream) {
  const int*   a1s = (const int*)d_in[0];
  const int*   a1d = (const int*)d_in[1];
  const float* a1w = (const float*)d_in[2];
  const int*   a2s = (const int*)d_in[3];
  const int*   a2d = (const int*)d_in[4];
  const float* a2w = (const float*)d_in[5];
  const float* W1  = (const float*)d_in[6];
  const float* biasv[6] = { (const float*)d_in[7],  (const float*)d_in[9],
                            (const float*)d_in[11], (const float*)d_in[13],
                            (const float*)d_in[15], (const float*)d_in[17] };
  const float* Wnext[5] = { (const float*)d_in[8],  (const float*)d_in[10],
                            (const float*)d_in[12], (const float*)d_in[14],
                            (const float*)d_in[16] };
  const float* L1  = (const float*)d_in[18];
  const float* bl1 = (const float*)d_in[19];
  const float* L2  = (const float*)d_in[20];
  const float* bl2 = (const float*)d_in[21];
  const float* L3  = (const float*)d_in[22];
  const float* bl3 = (const float*)d_in[23];

  const int E = in_sizes[0];
  const int N = in_sizes[6] / NHID;

  // workspace carve-out (256B aligned slices)
  char* ws = (char*)d_ws;
  size_t off = 0;
  auto carve = [&](size_t bytes) -> void* {
    void* p = ws + off;
    off = (off + bytes + 255) & ~(size_t)255;
    return p;
  };
  float*     y      = (float*)carve((size_t)N * NHID * 4);   // spmm output
  float*     xw     = (float*)carve((size_t)N * NHID * 4);   // x_i @ W_{i+1}
  float*     x7     = (float*)carve((size_t)N * NHID * 4);   // x1+..+x6 (f32)
  _Float16*  h2     = (_Float16*)carve((size_t)N * NH2 * 2); // mlp hidden 2
  _Float16*  Ah     = (_Float16*)carve((size_t)N * NHID * 2);// f16 activations
  float*     sacc   = (float*)carve((size_t)N * 4);          // mlp score accumulator
  int*       counts = (int*)carve((size_t)N * 4);
  int*       offs   = (int*)carve((size_t)(N + 1) * 4);
  int*       cursor = (int*)carve((size_t)N * 4);
  int*       csrs   = (int*)carve((size_t)E * 4);
  float*     csrw   = (float*)carve((size_t)E * 4);
  _Float16*  Wt[5];
  for (int i = 0; i < 5; i++) Wt[i] = (_Float16*)carve((size_t)NHID * NHID * 2);
  _Float16*  L1t = (_Float16*)carve((size_t)NH2 * NHID * 2);
  _Float16*  L2t = (_Float16*)carve((size_t)NH2 * NH2 * 2);
  _Float16*  h1  = (_Float16*)y;  // alias: N*256*f16 == N*128*f32; y is dead during MLP

  // ---- weight prep (tiny) ----
  for (int i = 0; i < 5; i++)
    k_cvt_transpose_f16<<<(NHID * NHID + 255) / 256, 256, 0, stream>>>(Wnext[i], Wt[i], NHID, NHID);
  k_cvt_transpose_f16<<<(NHID * NH2 + 255) / 256, 256, 0, stream>>>(L1, L1t, NHID, NH2);
  k_cvt_transpose_f16<<<(NH2 * NH2 + 255) / 256, 256, 0, stream>>>(L2, L2t, NH2, NH2);

  const int eblk = (E + 255) / 256;
  const int nblk = (N + 255) / 256;

  auto gemm = [&](const _Float16* A, const _Float16* Bt, int M, int K, int Nc,
                  const float* bias, int relu, float* Cf, _Float16* Ch) {
    long long waves = (long long)((M + 15) / 16) * (Nc / 64);  // 16x64 per wave
    int blocks = (int)((waves + 7) / 8);                       // 8 waves per block
    k_gemm_f16<<<blocks, 256, 0, stream>>>(A, Bt, M, K, Nc, bias, relu, Cf, Ch);
  };
  auto mlp_accum = [&](const _Float16* X) {
    gemm(X,  L1t, N, NHID, NH2, bl1, 1, nullptr, h1);
    gemm(h1, L2t, N, NH2,  NH2, bl2, 1, nullptr, h2);
    k_dot_l3<<<(N + 7) / 8, 256, 0, stream>>>(h2, L3, bl3, sacc, N);
  };

  auto branch = [&](const int* src, const int* dst, const float* w, int which) {
    // ---- CSR build for this adjacency (reused by all 6 SpMMs) ----
    hipMemsetAsync(counts, 0, (size_t)N * 4, stream);
    k_hist<<<eblk, 256, 0, stream>>>(dst, counts, E);
    k_scan<<<1, 1024, 0, stream>>>(counts, offs, N);
    k_copy_int<<<nblk, 256, 0, stream>>>(offs, cursor, N);
    k_scatter<<<eblk, 256, 0, stream>>>(src, dst, w, cursor, csrs, csrw, E);

    hipMemsetAsync(sacc, 0, (size_t)N * 4, stream);
    hipMemsetAsync(x7,   0, (size_t)N * NHID * 4, stream);
    const float* xin = W1;
    for (int i = 0; i < 6; i++) {
      k_spmm_csr<<<(N + 7) / 8, 256, 0, stream>>>(offs, csrs, csrw, xin, y, N);
      k_post<<<(N + 7) / 8, 256, 0, stream>>>(y, biasv[i], Ah, x7, N, (i < 5) ? 1 : 0);
      mlp_accum(Ah);
      if (i < 5) {
        gemm(Ah, Wt[i], N, NHID, NHID, nullptr, 0, xw, nullptr);
        xin = xw;
      }
    }
    size_t n7 = (size_t)N * NHID;
    k_cvt_f16<<<(int)((n7 + 255) / 256), 256, 0, stream>>>(x7, Ah, n7);
    mlp_accum(Ah);
    k_finalize<<<nblk, 256, 0, stream>>>(sacc, (float*)d_out, N, which);
  };

  branch(a1s, a1d, a1w, 0);
  branch(a2s, a2d, a2w, 1);
}
